// MultiHeadAttention_9216999817455
// MI455X (gfx1250) — matrix operand order repaired and checked by reading, not compiled, not run on hardware
//
#include <hip/hip_runtime.h>
#include <hip/hip_bf16.h>

typedef __attribute__((ext_vector_type(16))) __bf16 v16bf;
typedef __attribute__((ext_vector_type(8)))  __bf16 v8bf;
typedef __attribute__((ext_vector_type(8)))  float  v8f;

#define DMODEL 1024
#define NHEAD  16
#define DKH    64
#define SEQ    2048
#define BATCH  2
#define MTOT   (BATCH * SEQ)   // 4096 tokens

static __device__ __forceinline__ v8f wmma_bf16(v16bf a, v16bf b, v8f c) {
  // D = A(16x32 bf16) * B(32x16 bf16) + C(16x16 f32)
  return __builtin_amdgcn_wmma_f32_16x16x32_bf16(false, a, false, b, (short)0, c,
                                                 false, false);
}

// raw v_exp_f32 / v_rcp_f32 (no OCML subnormal guards -- softmax-safe)
static __device__ __forceinline__ float fast_exp2(float x) {
  return __builtin_amdgcn_exp2f(x);
}
static __device__ __forceinline__ float fast_rcp(float x) {
  return __builtin_amdgcn_rcpf(x);
}

// ---- operand loaders -------------------------------------------------------
// A 16x32 bf16: lane<16 -> row=lane, K 0..7 & 16..23; lane>=16 -> K 8..15 & 24..31
static __device__ __forceinline__ v16bf load_a_bf16(const __bf16* __restrict__ row,
                                                    int kb, int half) {
  v8bf lo = *(const v8bf*)(row + kb + half * 8);
  v8bf hi = *(const v8bf*)(row + kb + 16 + half * 8);
  v16bf r;
#pragma unroll
  for (int i = 0; i < 8; ++i) { r[i] = lo[i]; r[8 + i] = hi[i]; }
  return r;
}
// B 32x16 bf16: lane holds col N=lane%16; lanes 0-15 K 0..7 & 16..23, lanes 16-31 K 8..15 & 24..31
static __device__ __forceinline__ v16bf load_b_bf16(const __bf16* __restrict__ row,
                                                    int kb, int half) {
  v8bf lo = *(const v8bf*)(row + kb + half * 8);
  v8bf hi = *(const v8bf*)(row + kb + 16 + half * 8);
  v16bf r;
#pragma unroll
  for (int i = 0; i < 8; ++i) { r[i] = lo[i]; r[8 + i] = hi[i]; }
  return r;
}

// ---- async global->LDS staging (gfx1250, ASYNCcnt tracked) -----------------
static __device__ __forceinline__ void async_ld_b128(unsigned lds_off,
                                                     const void* gsrc) {
  asm volatile("global_load_async_to_lds_b128 %0, %1, off"
               :: "v"(lds_off), "v"((unsigned long long)(size_t)gsrc)
               : "memory");
}
static __device__ __forceinline__ void wait_async0() {
  asm volatile("s_wait_asynccnt 0x0" ::: "memory");
}

// ---- fp32 -> bf16 conversion prepass ---------------------------------------
__global__ void __launch_bounds__(256)
cvt_f32_bf16(const float* __restrict__ in, __bf16* __restrict__ out, int n) {
  int i = (blockIdx.x * blockDim.x + threadIdx.x) * 8;
  if (i >= n) return;
  const float* p = in + i;
  v8bf o;
#pragma unroll
  for (int j = 0; j < 8; ++j) o[j] = (__bf16)p[j];
  *(v8bf*)(out + i) = o;
}

// ---- GEMM: out[m,n] = (A[m,:] . W[n,:] + bias[n]) * oscale -----------------
// One wave computes a 32(M) x 64(N) tile: B tiles reused across two A tiles.
// MODE 0: bf16 row-major [MTOT, DMODEL]
// MODE 1: bf16 V-transposed [BATCH, NHEAD, DKH, SEQ]
// MODE 2: fp32 row-major (final output)
template <int MODE>
__global__ void __launch_bounds__(256)
gemm_bf16(const __bf16* __restrict__ A, const __bf16* __restrict__ W,
          const float* __restrict__ bias, void* __restrict__ outv, float oscale) {
  const int lane = threadIdx.x & 31;
  const int half = lane >> 4;
  const int wid  = (blockIdx.x * blockDim.x + threadIdx.x) >> 5;
  const int TN = DMODEL / 64;                 // 16 n-tiles
  const int m0 = (wid / TN) * 32;
  const int n0 = (wid % TN) * 64;

  const __bf16* arow0 = A + (size_t)(m0 + (lane & 15)) * DMODEL;
  const __bf16* arow1 = arow0 + (size_t)16 * DMODEL;
  const __bf16* wrow[4];
  float bcol[4];
#pragma unroll
  for (int t = 0; t < 4; ++t) {
    int n = n0 + t * 16 + (lane & 15);
    wrow[t] = W + (size_t)n * DMODEL;
    bcol[t] = bias[n];
  }

  v8f acc[2][4] = {};
  for (int kb = 0; kb < DMODEL; kb += 32) {
    v16bf a0 = load_a_bf16(arow0, kb, half);
    v16bf a1 = load_a_bf16(arow1, kb, half);
#pragma unroll
    for (int t = 0; t < 4; ++t) {
      v16bf bm = load_b_bf16(wrow[t], kb, half);
      acc[0][t] = wmma_bf16(a0, bm, acc[0][t]);
      acc[1][t] = wmma_bf16(a1, bm, acc[1][t]);
    }
  }

  // C/D layout: lane n = lane%16, VGPR j holds row m = j + 8*(lane/16)
#pragma unroll
  for (int rr = 0; rr < 2; ++rr) {
#pragma unroll
    for (int t = 0; t < 4; ++t) {
      int n = n0 + t * 16 + (lane & 15);
      if constexpr (MODE == 2) {
        float* out = (float*)outv;
#pragma unroll
        for (int j = 0; j < 8; ++j) {
          int m = m0 + rr * 16 + j + half * 8;
          out[(size_t)m * DMODEL + n] = (acc[rr][t][j] + bcol[t]) * oscale;
        }
      } else if constexpr (MODE == 0) {
        __bf16* out = (__bf16*)outv;
#pragma unroll
        for (int j = 0; j < 8; ++j) {
          int m = m0 + rr * 16 + j + half * 8;
          out[(size_t)m * DMODEL + n] = (__bf16)((acc[rr][t][j] + bcol[t]) * oscale);
        }
      } else {  // MODE 1: Vt[b][h][dk][s], per-lane contiguous in s
        __bf16* out = (__bf16*)outv;
        int h = n >> 6, dk = n & 63;
        int mbase = m0 + rr * 16 + half * 8;
        int bb = mbase >> 11, s = mbase & (SEQ - 1);
        __bf16* dst = out + (((size_t)(bb * NHEAD + h) * DKH + dk) * SEQ + s);
#pragma unroll
        for (int j = 0; j < 8; ++j) dst[j] = (__bf16)((acc[rr][t][j] + bcol[t]) * oscale);
      }
    }
  }
}

// ---- flash attention -------------------------------------------------------
// Block = 4 waves = 64 consecutive queries of one (b,h). K/V tiles (32 keys)
// staged once per block into LDS via async DMA, double-buffered.
// Scores computed TRANSPOSED: S^T = K . Q^T, so each lane owns one query's
// scores (softmax is lane-local) and the bf16-converted S^T accumulators ARE
// the A-operand layout of P for the P.V WMMA -- no LDS transpose needed.
// Q is pre-scaled by log2(e)/sqrt(dk), so softmax uses native v_exp_f32.
__global__ void __launch_bounds__(128)
flash_attn(const __bf16* __restrict__ Qp, const __bf16* __restrict__ Kp,
           const __bf16* __restrict__ Vt, __bf16* __restrict__ Ctx) {
  __shared__ __attribute__((aligned(32))) __bf16 Klds[2][32 * 72];  // 32 keys x 64 dk, padded
  __shared__ __attribute__((aligned(32))) __bf16 Vlds[2][64 * 40];  // 64 dk x 32 keys, padded

  const int tid   = threadIdx.x;
  const int lane  = tid & 31;
  const int half  = lane >> 4;
  const int qlane = lane & 15;
  const int wid   = (blockIdx.x * blockDim.x + tid) >> 5;
  const int QT = SEQ / 16;                  // 128 q-tiles per head
  const int b  = wid / (NHEAD * QT);
  const int r  = wid % (NHEAD * QT);
  const int h  = r / QT;
  const int q0 = (r % QT) * 16;

  // Q as B-operand (col = query, contraction = dk, contiguous per lane)
  const __bf16* qrow = Qp + (size_t)(b * SEQ + q0 + qlane) * DMODEL + h * DKH;
  v16bf qb0 = load_b_bf16(qrow, 0, half);   // dk 0..31
  v16bf qb1 = load_b_bf16(qrow, 32, half);  // dk 32..63

  const __bf16* kbase = Kp + (size_t)b * SEQ * DMODEL + h * DKH;
  const __bf16* vbase = Vt + (size_t)(b * NHEAD + h) * DKH * SEQ;

  // cooperative async staging of one 32-key K/V tile
  auto stage = [&](int kv, __bf16* kl, __bf16* vl) {
    int krow = tid >> 3, kcol = (tid & 7) * 8;      // K: 32 x 64
#pragma unroll
    for (int i = 0; i < 2; ++i) {
      int rr = krow + 16 * i;
      async_ld_b128((unsigned)(size_t)&kl[rr * 72 + kcol],
                    kbase + (size_t)(kv + rr) * DMODEL + kcol);
    }
    int vrow = tid >> 2, vcol = (tid & 3) * 8;      // V: 64 x 32
#pragma unroll
    for (int i = 0; i < 2; ++i) {
      int rr = vrow + 32 * i;
      async_ld_b128((unsigned)(size_t)&vl[rr * 40 + vcol],
                    vbase + (size_t)rr * SEQ + kv + vcol);
    }
  };

  float mst = -1e30f, lst = 0.0f;           // per-lane (= per-query) stats
  v8f oacc[4] = {};

  stage(0, &Klds[0][0], &Vlds[0][0]);
  wait_async0();
  __syncthreads();

  for (int kv = 0; kv < SEQ; kv += 32) {
    const int cur = (kv >> 5) & 1;
    if (kv + 32 < SEQ)
      stage(kv + 32, &Klds[cur ^ 1][0], &Vlds[cur ^ 1][0]);  // overlap DMA w/ math

    const __bf16* kl = &Klds[cur][0];
    const __bf16* vl = &Vlds[cur][0];

    // S^T(32k x 16q): two 16x16 tiles; A = K tile rows (keys), B = Q^T
    v16bf ka[2][2];
#pragma unroll
    for (int t = 0; t < 2; ++t) {
      const __bf16* krow = kl + (size_t)(t * 16 + qlane) * 72;
      ka[t][0] = load_a_bf16(krow, 0, half);
      ka[t][1] = load_a_bf16(krow, 32, half);
    }
    v8f st[2];
#pragma unroll
    for (int t = 0; t < 2; ++t) {
      v8f z = {};
      z = wmma_bf16(ka[t][0], qb0, z);
      z = wmma_bf16(ka[t][1], qb1, z);
      st[t] = z;   // lane holds query qlane; reg j = key t*16 + j + 8*half
    }

    // lane-local softmax over this lane's 16 scores (+ partner half via xor16)
    float mx = st[0][0];
#pragma unroll
    for (int j = 1; j < 8; ++j) mx = fmaxf(mx, st[0][j]);
#pragma unroll
    for (int j = 0; j < 8; ++j) mx = fmaxf(mx, st[1][j]);
    mx = fmaxf(mx, __shfl_xor(mx, 16, 32));
    float mnew  = fmaxf(mst, mx);
    float alpha = fast_exp2(mst - mnew);
    mst = mnew;

    float ls = 0.0f;
#pragma unroll
    for (int j = 0; j < 8; ++j) {
      float p0 = fast_exp2(st[0][j] - mnew);
      float p1 = fast_exp2(st[1][j] - mnew);
      st[0][j] = p0; st[1][j] = p1;
      ls += p0 + p1;
    }
    ls += __shfl_xor(ls, 16, 32);
    lst = lst * alpha + ls;

    // P in A-operand layout directly from the transposed accumulators
    v16bf pa;
#pragma unroll
    for (int j = 0; j < 8; ++j) {
      pa[j]     = (__bf16)st[0][j];   // keys half*8 + 0..7
      pa[8 + j] = (__bf16)st[1][j];   // keys 16 + half*8 + 0..7
    }

    // broadcast per-query alpha into O's row layout and rescale
#pragma unroll
    for (int j = 0; j < 8; ++j) {
      float aj = __shfl(alpha, j + 8 * half, 32);
#pragma unroll
      for (int t = 0; t < 4; ++t) oacc[t][j] *= aj;
    }

    // O(16q x 64dk) += P(16x32) . V(32x64); V rows (dk) have keys contiguous
    v16bf vb[4];
#pragma unroll
    for (int t = 0; t < 4; ++t)
      vb[t] = load_b_bf16(vl + (size_t)(t * 16 + qlane) * 40, 0, half);
#pragma unroll
    for (int t = 0; t < 4; ++t)
      oacc[t] = wmma_bf16(pa, vb[t], oacc[t]);

    wait_async0();        // next tile's DMA landed
    __syncthreads();
  }

  // normalize (per-query 1/l broadcast into row layout) and store ctx
  float rinv[8];
#pragma unroll
  for (int j = 0; j < 8; ++j) rinv[j] = fast_rcp(__shfl(lst, j + 8 * half, 32));
#pragma unroll
  for (int t = 0; t < 4; ++t) {
    int n = h * DKH + t * 16 + qlane;
#pragma unroll
    for (int j = 0; j < 8; ++j) {
      int m = b * SEQ + q0 + j + half * 8;
      Ctx[(size_t)m * DMODEL + n] = (__bf16)(oacc[t][j] * rinv[j]);
    }
  }
}

extern "C" void kernel_launch(void* const* d_in, const int* in_sizes, int n_in,
                              void* d_out, int out_size, void* d_ws, size_t ws_size,
                              hipStream_t stream) {
  const float* q  = (const float*)d_in[0];
  const float* k  = (const float*)d_in[1];
  const float* v  = (const float*)d_in[2];
  const float* Wq = (const float*)d_in[3];
  const float* bq = (const float*)d_in[4];
  const float* Wk = (const float*)d_in[5];
  const float* bk = (const float*)d_in[6];
  const float* Wv = (const float*)d_in[7];
  const float* bv = (const float*)d_in[8];
  const float* Wo = (const float*)d_in[9];
  const float* bo = (const float*)d_in[10];
  float* out = (float*)d_out;

  char* ws = (char*)d_ws;
  const size_t ACT = (size_t)MTOT * DMODEL * sizeof(__bf16);    // 8 MB
  const size_t WSZ = (size_t)DMODEL * DMODEL * sizeof(__bf16);  // 2 MB
  __bf16* Xq  = (__bf16*)(ws);
  __bf16* Xk  = (__bf16*)(ws + ACT);
  __bf16* Xv  = (__bf16*)(ws + 2 * ACT);
  __bf16* Wqb = (__bf16*)(ws + 3 * ACT);
  __bf16* Wkb = (__bf16*)(ws + 3 * ACT + WSZ);
  __bf16* Wvb = (__bf16*)(ws + 3 * ACT + 2 * WSZ);
  __bf16* Wob = (__bf16*)(ws + 3 * ACT + 3 * WSZ);
  __bf16* Qp  = (__bf16*)(ws + 3 * ACT + 4 * WSZ);
  __bf16* Kp  = (__bf16*)(ws + 4 * ACT + 4 * WSZ);
  __bf16* Vtp = (__bf16*)(ws + 5 * ACT + 4 * WSZ);   // [B, H, DKH, SEQ]
  __bf16* Ctx = (__bf16*)(ws + 6 * ACT + 4 * WSZ);

  const int NA = MTOT * DMODEL;     // 4 Mi elements
  const int NW = DMODEL * DMODEL;   // 1 Mi elements
  cvt_f32_bf16<<<dim3(NA / 8 / 256), dim3(256), 0, stream>>>(q,  Xq,  NA);
  cvt_f32_bf16<<<dim3(NA / 8 / 256), dim3(256), 0, stream>>>(k,  Xk,  NA);
  cvt_f32_bf16<<<dim3(NA / 8 / 256), dim3(256), 0, stream>>>(v,  Xv,  NA);
  cvt_f32_bf16<<<dim3(NW / 8 / 256), dim3(256), 0, stream>>>(Wq, Wqb, NW);
  cvt_f32_bf16<<<dim3(NW / 8 / 256), dim3(256), 0, stream>>>(Wk, Wkb, NW);
  cvt_f32_bf16<<<dim3(NW / 8 / 256), dim3(256), 0, stream>>>(Wv, Wvb, NW);
  cvt_f32_bf16<<<dim3(NW / 8 / 256), dim3(256), 0, stream>>>(Wo, Wob, NW);

  // log2(e)/sqrt(dk) folded into Q so flash softmax uses native exp2
  const float QSCALE = 1.4426950408889634f * 0.125f;

  // 2048 wave-tiles (32x64 each) -> 256 blocks x 256 threads (8 waves, wave32)
  gemm_bf16<0><<<dim3(256), dim3(256), 0, stream>>>(Xq, Wqb, bq, Qp, QSCALE);
  gemm_bf16<0><<<dim3(256), dim3(256), 0, stream>>>(Xk, Wkb, bk, Kp, 1.0f);
  gemm_bf16<1><<<dim3(256), dim3(256), 0, stream>>>(Xv, Wvb, bv, Vtp, 1.0f);
  // 4096 (b,h,qtile) waves -> 1024 blocks x 128 threads (4 waves/block)
  flash_attn<<<dim3(1024), dim3(128), 0, stream>>>(Qp, Kp, Vtp, Ctx);
  gemm_bf16<2><<<dim3(256), dim3(256), 0, stream>>>(Ctx, Wob, bo, out, 1.0f);
}
